// Mamba_Diffusion_UViT_68822555951851
// MI455X (gfx1250) — compile-verified, matrix operand order checked by reading, not run on hardware
//
#include <hip/hip_runtime.h>
#include <hip/hip_bf16.h>
#include <math.h>

// ---------------- CDNA5 WMMA types ----------------
typedef __attribute__((ext_vector_type(16))) __bf16 v16bf;
typedef __attribute__((ext_vector_type(2)))  __bf16 v2bf;
typedef __attribute__((ext_vector_type(2)))  float  v2f;
typedef __attribute__((ext_vector_type(8)))  float  v8f;

#define DEVI static __device__ __forceinline__

// native packed convert: lowers to v_cvt_pk_bf16_f32 on gfx1250
DEVI unsigned pack2(float a, float b) {
  v2f f; f.x = a; f.y = b;
  union { v2bf h; unsigned u; } c;
  c.h = __builtin_convertvector(f, v2bf);
  return c.u;
}
DEVI float siluf(float x)     { return x / (1.f + __expf(-x)); }
DEVI float softplusf(float x) { return (x > 20.f) ? x : log1pf(__expf(x)); }

union FragB { v16bf v; unsigned u[8]; };

// LDS layouts (dwords):
//   sa: [row][kpair]                       (16 dwords/row; fragment = 2x b128)
//   sb: [n][(kp>>2) ^ ((n>>2)&3)][kp&3]    (16 dwords/n; XOR-swizzled kpair groups
//        so staging b128 stores are ~conflict-free AND fragments = 2x b128)
template<bool GW>
DEVI void stage_tiles(unsigned* __restrict__ sa, unsigned* __restrict__ sb,
                      const float* __restrict__ A, int lda,
                      const float* __restrict__ W, int N, int Kw,
                      int tm, int tn, int k0, int tid) {
  const int r0 = tid >> 3, q8 = tid & 7;     // A mapping: row base, float4-in-row
  // ---- issue A loads (4 x b128, unconditional: rows padded, K%32==0) ----
  float4 fa[4];
  #pragma unroll
  for (int i = 0; i < 4; i++)
    fa[i] = *(const float4*)(A + (size_t)(tm + i * 32 + r0) * lda + k0 + (q8 << 2));

  if (!GW) {
    // ---- B: thread owns (column n, k-group g) -> 8 coalesced b32 loads ----
    const int n = tid & 63;
    const int g = tid >> 6;                  // kpair group 0..3 (k = 8g..8g+7)
    const float* wp = W + (size_t)(k0 + 8 * g) * N + tn + n;
    float r[8];
    #pragma unroll
    for (int j = 0; j < 8; j++) r[j] = wp[(size_t)j * N];
    // ---- pack + LDS stores ----
    #pragma unroll
    for (int i = 0; i < 4; i++) {
      int base = (i * 32 + r0) * 16 + (q8 << 1);
      sa[base    ] = pack2(fa[i].x, fa[i].y);
      sa[base + 1] = pack2(fa[i].z, fa[i].w);
    }
    uint4 uv;
    uv.x = pack2(r[0], r[1]);
    uv.y = pack2(r[2], r[3]);
    uv.z = pack2(r[4], r[5]);
    uv.w = pack2(r[6], r[7]);
    *(uint4*)&sb[(n << 4) + ((g ^ ((n >> 2) & 3)) << 2)] = uv;  // swizzled b128 store
  } else {
    // ---- guarded B loads (clamped addresses, select-zero, no divergence) ----
    float g0[4], g1[4];
    #pragma unroll
    for (int i = 0; i < 4; i++) {
      int idx = i * 256 + tid;               // 0..1023 : (kpair kk, column n)
      int kk = idx >> 6, n = idx & 63;
      int gk = k0 + 2 * kk, gn = tn + n;
      int gka = (gk     < Kw) ? gk     : (Kw - 1);
      int gkb = (gk + 1 < Kw) ? gk + 1 : (Kw - 1);
      int gnc = (gn < N) ? gn : (N - 1);
      g0[i] = W[(size_t)gka * N + gnc];
      g1[i] = W[(size_t)gkb * N + gnc];
    }
    #pragma unroll
    for (int i = 0; i < 4; i++) {
      int base = (i * 32 + r0) * 16 + (q8 << 1);
      sa[base    ] = pack2(fa[i].x, fa[i].y);
      sa[base + 1] = pack2(fa[i].z, fa[i].w);
    }
    #pragma unroll
    for (int i = 0; i < 4; i++) {
      int idx = i * 256 + tid;
      int kk = idx >> 6, n = idx & 63;
      int gk = k0 + 2 * kk, gn = tn + n;
      float f0 = (gk     >= Kw || gn >= N) ? 0.f : g0[i];
      float f1 = (gk + 1 >= Kw || gn >= N) ? 0.f : g1[i];
      sb[(n << 4) + (((kk >> 2) ^ ((n >> 2) & 3)) << 2) + (kk & 3)] = pack2(f0, f1);
    }
  }
}

// One K-step: A fragment (2x ds_load_b128) + 4 B fragments (2x ds_load_b128 each), 4 WMMAs.
// Pointers must remain LDS address space (callers index the __shared__ arrays directly).
DEVI void wmma_step(const unsigned* aRow, const unsigned* bBuf,
                    int half, int l15, v8f* __restrict__ acc) {
  FragB af;
  {
    const uint4 a0 = *(const uint4*)(aRow + (half << 2));
    const uint4 a1 = *(const uint4*)(aRow + 8 + (half << 2));
    af.u[0] = a0.x; af.u[1] = a0.y; af.u[2] = a0.z; af.u[3] = a0.w;
    af.u[4] = a1.x; af.u[5] = a1.y; af.u[6] = a1.z; af.u[7] = a1.w;
  }
  const int s = l15 >> 2;                    // == (n>>2)&3 for this lane, any nt
  #pragma unroll
  for (int nt = 0; nt < 4; nt++) {
    const unsigned* bn = bBuf + (((nt << 4) + l15) << 4);
    const uint4 lo = *(const uint4*)(bn + ((half ^ s) << 2));
    const uint4 hi = *(const uint4*)(bn + (((2 + half) ^ s) << 2));
    FragB bf;
    bf.u[0] = lo.x; bf.u[1] = lo.y; bf.u[2] = lo.z; bf.u[3] = lo.w;
    bf.u[4] = hi.x; bf.u[5] = hi.y; bf.u[6] = hi.z; bf.u[7] = hi.w;
    acc[nt] = __builtin_amdgcn_wmma_f32_16x16x32_bf16(
        false, af.v, false, bf.v, (short)0, acc[nt], false, false);
  }
}

// ---------------- Generic fused GEMM ----------------
// C = act(A@W + bias) [+resid (ACT0) | *aux (ACT3)]
// 128x64 tile / workgroup, 8 waves x (16x64), LDS double-buffered, last iter peeled.
template<int ACT, bool RES, bool GW>
__global__ __launch_bounds__(256)
void gemm_wmma_kernel(const float* __restrict__ A, int lda,
                      const float* __restrict__ W,
                      const float* __restrict__ bias,
                      const float* __restrict__ resid, int ldr,
                      float* __restrict__ C, int ldc,
                      int N, int K, int Kw)
{
  __shared__ unsigned sa[2][128 * 16];
  __shared__ unsigned sb[2][64 * 16];
  const int tid  = threadIdx.x;
  const int wave = tid >> 5;
  const int lane = tid & 31;
  const int half = lane >> 4;
  const int l15  = lane & 15;
  const int tm = blockIdx.y * 128;
  const int tn = blockIdx.x * 64;
  const int aoff = ((wave << 4) + l15) << 4;   // this lane's A-row base (dwords)

  v8f acc[4];
  #pragma unroll
  for (int i = 0; i < 4; i++)
    #pragma unroll
    for (int j = 0; j < 8; j++) acc[i][j] = 0.f;

  stage_tiles<GW>(sa[0], sb[0], A, lda, W, N, Kw, tm, tn, 0, tid);
  __syncthreads();

  int cur = 0, k0 = 0;
  // steady state: compute(cur) overlapped with stage(next); one barrier per step.
  // NOTE: sa[cur]/sb[cur] direct indexing keeps LDS addrspace => ds_load_b128.
  for (; k0 + 32 < K; k0 += 32) {
    wmma_step(sa[cur] + aoff, sb[cur], half, l15, acc);
    stage_tiles<GW>(sa[cur ^ 1], sb[cur ^ 1], A, lda, W, N, Kw, tm, tn, k0 + 32, tid);
    if (!GW) {  // branchless clamped prefetch of the K+64 weight slab
      int kpf = (k0 + 64 < K) ? (k0 + 64) : 0;
      __builtin_prefetch(W + (size_t)kpf * N + tn, 0, 1);
    }
    __syncthreads();
    cur ^= 1;
  }
  wmma_step(sa[cur] + aoff, sb[cur], half, l15, acc);   // peeled last K-step

  // Epilogue: D layout — lanes 0-15: N=lane, vgpr v -> M=v; lanes 16-31: M=v+8
  #pragma unroll
  for (int nt = 0; nt < 4; nt++) {
    int gn = tn + (nt << 4) + l15;
    if (GW && gn >= N) continue;
    float bv = bias ? bias[gn] : 0.f;
    #pragma unroll
    for (int v = 0; v < 8; v++) {
      int gm = tm + (wave << 4) + v + (half << 3);
      float x = acc[nt][v] + bv;
      if (ACT == 1) x = siluf(x);
      else if (ACT == 2) x = softplusf(x);
      else if (ACT == 3) x = siluf(x) * resid[(size_t)gm * ldr + gn];
      if (RES) x += resid[(size_t)gm * ldr + gn];
      C[(size_t)gm * ldc + gn] = x;
    }
  }
}

// ---------------- Elementwise / special kernels ----------------
__global__ void time_embed_kernel(const float* __restrict__ ts, float* __restrict__ emb) {
  int i = blockIdx.x * blockDim.x + threadIdx.x;   // 4*1024
  if (i >= 4 * 1024) return;
  int b = i >> 10, d = i & 1023;
  int j = d & 511;
  float fr = __expf(-logf(10000.f) * (float)j / 512.f);
  float a = ts[b] * fr;
  emb[i] = (d < 512) ? __cosf(a) : __sinf(a);
}

__global__ void patchify_kernel(const float* __restrict__ hs, float* __restrict__ P) {
  int i = blockIdx.x * blockDim.x + threadIdx.x;   // 4*256*16
  if (i >= 4 * 256 * 16) return;
  int e = i & 15, tok = (i >> 4) & 255, b = i >> 12;
  int c = e >> 2, p1 = (e >> 1) & 1, p2 = e & 1;
  int h1 = tok >> 4, w1 = tok & 15;
  // PP has lda=32 (cols 16..31 zero-padded so the patch GEMM can run K=32)
  P[(size_t)((b << 8) + tok) * 32 + e] =
      hs[((b * 4 + c) * 32 + (h1 * 2 + p1)) * 32 + (w1 * 2 + p2)];
}

__global__ void assemble_kernel(const float* __restrict__ PT, const float* __restrict__ TT,
                                const float* __restrict__ pos, float* __restrict__ X) {
  int i = blockIdx.x * blockDim.x + threadIdx.x;   // 4*257*1024
  if (i >= 4 * 257 * 1024) return;
  int d = i & 1023;
  int t = (i >> 10) % 257;
  int b = i / (257 * 1024);
  float v = (t == 0) ? TT[b * 1024 + d] : PT[(size_t)((b << 8) + (t - 1)) * 1024 + d];
  X[(size_t)(b * 257 + t) * 1024 + d] = v + pos[t * 1024 + d];
}

__global__ __launch_bounds__(256)
void layernorm_kernel(const float* __restrict__ X, float* __restrict__ Y,
                      const float* __restrict__ gamma, const float* __restrict__ beta) {
  __shared__ float red[256];
  int row = blockIdx.x, tid = threadIdx.x;
  const float* xr = X + (size_t)row * 1024;
  float s = 0.f;
  for (int i = tid; i < 1024; i += 256) s += xr[i];
  red[tid] = s; __syncthreads();
  for (int o = 128; o > 0; o >>= 1) { if (tid < o) red[tid] += red[tid + o]; __syncthreads(); }
  float mean = red[0] * (1.f / 1024.f); __syncthreads();
  float v = 0.f;
  for (int i = tid; i < 1024; i += 256) { float d = xr[i] - mean; v += d * d; }
  red[tid] = v; __syncthreads();
  for (int o = 128; o > 0; o >>= 1) { if (tid < o) red[tid] += red[tid + o]; __syncthreads(); }
  float inv = rsqrtf(red[0] * (1.f / 1024.f) + 1e-6f);
  float* yr = Y + (size_t)row * 1024;
  for (int i = tid; i < 1024; i += 256) {
    float o2 = (xr[i] - mean) * inv;
    if (gamma) o2 = o2 * gamma[i] + beta[i];
    yr[i] = o2;
  }
}

// depthwise causal conv K=4 over token dim + SiLU; xm = XZ[:, :2048] (row stride 4096)
__global__ void dwconv_silu_kernel(const float* __restrict__ XZ, const float* __restrict__ cw,
                                   const float* __restrict__ cb, float* __restrict__ XC) {
  int i = blockIdx.x * blockDim.x + threadIdx.x;   // 4*257*2048
  if (i >= 4 * 257 * 2048) return;
  int c = i & 2047;
  int t = (i >> 11) % 257;
  int b = i / (257 * 2048);
  float acc = cb[c];
  #pragma unroll
  for (int k = 0; k < 4; k++) {
    int tk = t + k - 3;
    if (tk >= 0) acc += XZ[(size_t)(b * 257 + tk) * 4096 + c] * cw[c * 4 + k];
  }
  XC[(size_t)(b * 257 + t) * 2048 + c] = siluf(acc);
}

// selective scan: parallel over (b, channel), sequential over L=257; B/C staged in LDS
__global__ __launch_bounds__(256)
void mamba_scan_kernel(const float* __restrict__ DELTA, const float* __restrict__ DBC,
                       const float* __restrict__ XC, const float* __restrict__ XZ,
                       const float* __restrict__ A_log, const float* __restrict__ Dp,
                       float* __restrict__ Y) {
  __shared__ float Bs[16], Cs[16];
  int b = blockIdx.y;
  int c = blockIdx.x * 256 + threadIdx.x;          // 0..2047
  float Aa[16], h[16];
  #pragma unroll
  for (int n = 0; n < 16; n++) { Aa[n] = -__expf(A_log[c * 16 + n]); h[n] = 0.f; }
  float dp = Dp[c];
  for (int t = 0; t < 257; t++) {
    size_t row = (size_t)b * 257 + t;
    __syncthreads();
    if (threadIdx.x < 16)       Bs[threadIdx.x]      = DBC[row * 96 + 64 + threadIdx.x];
    else if (threadIdx.x < 32)  Cs[threadIdx.x - 16] = DBC[row * 96 + 80 + (threadIdx.x - 16)];
    __syncthreads();
    float d  = DELTA[row * 2048 + c];
    float u  = XC[row * 2048 + c];
    float du = d * u;
    float y  = 0.f;
    #pragma unroll
    for (int n = 0; n < 16; n++) {
      h[n] = h[n] * __expf(d * Aa[n]) + du * Bs[n];
      y += h[n] * Cs[n];
    }
    y += u * dp;
    float z = XZ[row * 4096 + 2048 + c];
    Y[row * 2048 + c] = y * siluf(z);
  }
}

// unpatchify + 3x3 SAME conv + bias
__global__ void final_head_kernel(const float* __restrict__ DEC, const float* __restrict__ fw,
                                  const float* __restrict__ fb, float* __restrict__ out) {
  int i = blockIdx.x * blockDim.x + threadIdx.x;   // 4*4*32*32
  if (i >= 16384) return;
  int x = i & 31, y = (i >> 5) & 31, c = (i >> 10) & 3, b = i >> 12;
  float acc = fb[c];
  for (int ci = 0; ci < 4; ci++)
    for (int kh = 0; kh < 3; kh++) {
      int yy = y + kh - 1; if (yy < 0 || yy > 31) continue;
      for (int kw = 0; kw < 3; kw++) {
        int xx = x + kw - 1; if (xx < 0 || xx > 31) continue;
        float v = DEC[((size_t)b * 257 + 1 + (yy >> 1) * 16 + (xx >> 1)) * 16
                      + ((yy & 1) * 8 + (xx & 1) * 4 + ci)];
        acc += v * fw[((c * 4 + ci) * 3 + kh) * 3 + kw];
      }
    }
  out[i] = acc;
}

// ---------------- host orchestration ----------------
static void launch_gemm(int act, bool res, bool gw,
                        const float* A, int lda, const float* W, const float* bias,
                        const float* resid, int ldr, float* C, int ldc,
                        int Mpad, int N, int K, int Kw, hipStream_t s) {
  dim3 g((N + 63) / 64, Mpad / 128), blk(256);
  #define GL(a, r, w) gemm_wmma_kernel<a, r, w><<<g, blk, 0, s>>>(A, lda, W, bias, resid, ldr, C, ldc, N, K, Kw)
  if (gw)            GL(0, false, true);
  else if (res)      GL(0, true,  false);
  else if (act == 0) GL(0, false, false);
  else if (act == 1) GL(1, false, false);
  else if (act == 2) GL(2, false, false);
  else               GL(3, false, false);
  #undef GL
}

extern "C" void kernel_launch(void* const* d_in, const int* in_sizes, int n_in,
                              void* d_out, int out_size, void* d_ws, size_t ws_size,
                              hipStream_t stream) {
  (void)in_sizes; (void)n_in; (void)out_size; (void)ws_size;
  // jax pytree (sorted-key) flatten order
  const float* hs      = (const float*)d_in[0];
  const float* beta    = (const float*)d_in[1];
  const float* A_log   = (const float*)d_in[2];
  const float* Dp      = (const float*)d_in[3];
  const float* conv_b  = (const float*)d_in[4];
  const float* conv_w  = (const float*)d_in[5];
  const float* dW1     = (const float*)d_in[6];
  const float* dW2     = (const float*)d_in[7];
  const float* db1     = (const float*)d_in[8];
  const float* db2     = (const float*)d_in[9];
  const float* dt_b    = (const float*)d_in[10];
  const float* dt_w    = (const float*)d_in[11];
  const float* fWf     = (const float*)d_in[12];
  const float* fWm     = (const float*)d_in[13];
  const float* fWr     = (const float*)d_in[14];
  const float* fbf     = (const float*)d_in[15];
  const float* fbm     = (const float*)d_in[16];
  const float* fbr     = (const float*)d_in[17];
  const float* in_w    = (const float*)d_in[18];
  const float* out_w   = (const float*)d_in[19];
  const float* xproj_w = (const float*)d_in[20];
  const float* dec_b   = (const float*)d_in[21];
  const float* dec_w   = (const float*)d_in[22];
  const float* fconv_b = (const float*)d_in[23];
  const float* fconv_w = (const float*)d_in[24];
  const float* gamma   = (const float*)d_in[25];
  const float* patch_b = (const float*)d_in[26];
  const float* patch_w = (const float*)d_in[27];
  const float* pos     = (const float*)d_in[28];
  const float* tb1     = (const float*)d_in[29];
  const float* tb2     = (const float*)d_in[30];
  const float* tw1     = (const float*)d_in[31];
  const float* tw2     = (const float*)d_in[32];
  const float* ts      = (const float*)d_in[33];
  float* out = (float*)d_out;

  const int MT = 4 * 257;   // 1028 real token rows
  const int MP = 1152;      // padded to 9 x 128-row tiles (no M guards anywhere)
  float* ws = (float*)d_ws;
  size_t off = 0;
  auto alloc = [&](size_t n) { float* p = ws + off; off += n; return p; };
  float* X     = alloc((size_t)MP * 1024);
  float* X1    = alloc((size_t)MP * 1024);
  float* H1    = alloc((size_t)MP * 512);
  float* HD    = alloc((size_t)MP * 1024);
  float* XZ    = alloc((size_t)MP * 4096);
  float* XC    = alloc((size_t)MP * 2048);
  float* DBC   = alloc((size_t)MP * 96);
  float* DELTA = alloc((size_t)MP * 2048);
  float* Yb    = alloc((size_t)MP * 2048);
  float* MO    = alloc((size_t)MP * 1024);
  float* AO    = alloc((size_t)MP * 512);
  float* RO    = alloc((size_t)MP * 512);
  float* EMB   = alloc((size_t)128 * 1024);
  float* T1    = alloc((size_t)128 * 4096);
  float* TT    = alloc((size_t)128 * 1024);
  float* PP    = alloc((size_t)1024 * 32);   // lda=32, cols 16..31 zeroed
  float* PT    = alloc((size_t)1024 * 1024);
  float* XF    = alloc((size_t)MP * 1024);
  float* DEC   = alloc((size_t)MP * 16);

  auto grd = [](int n) { return dim3((n + 255) / 256); };

  // --- stem ---
  hipMemsetAsync(PP, 0, (size_t)1024 * 32 * sizeof(float), stream);
  time_embed_kernel<<<grd(4 * 1024), 256, 0, stream>>>(ts, EMB);
  launch_gemm(1, false, false, EMB, 1024, tw1, tb1, nullptr, 0, T1, 4096, 128, 4096, 1024, 1024, stream);
  launch_gemm(0, false, false, T1, 4096, tw2, tb2, nullptr, 0, TT, 1024, 128, 1024, 4096, 4096, stream);
  patchify_kernel<<<grd(4 * 256 * 16), 256, 0, stream>>>(hs, PP);
  launch_gemm(0, false, true, PP, 32, patch_w, patch_b, nullptr, 0, PT, 1024, 1024, 1024, 32, 16, stream);
  assemble_kernel<<<grd(4 * 257 * 1024), 256, 0, stream>>>(PT, TT, pos, X);

  // --- 13 mamba blocks ---
  for (int nb = 0; nb < 13; nb++) {
    const float* bdW1 = dW1  + (size_t)nb * 1024 * 512;
    const float* bdb1 = db1  + (size_t)nb * 512;
    const float* bdW2 = dW2  + (size_t)nb * 512 * 1024;
    const float* bdb2 = db2  + (size_t)nb * 1024;
    const float* binw = in_w + (size_t)nb * 1024 * 4096;
    const float* bcw  = conv_w + (size_t)nb * 2048 * 4;
    const float* bcb  = conv_b + (size_t)nb * 2048;
    const float* bxp  = xproj_w + (size_t)nb * 2048 * 96;
    const float* bdtw = dt_w + (size_t)nb * 64 * 2048;
    const float* bdtb = dt_b + (size_t)nb * 2048;
    const float* bAl  = A_log + (size_t)nb * 2048 * 16;
    const float* bDp  = Dp + (size_t)nb * 2048;
    const float* bow  = out_w + (size_t)nb * 2048 * 1024;
    const float* bfWm = fWm + (size_t)nb * 1024 * 512;
    const float* bfbm = fbm + (size_t)nb * 512;
    const float* bfWr = fWr + (size_t)nb * 1024 * 512;
    const float* bfbr = fbr + (size_t)nb * 512;
    const float* bfWf = fWf + (size_t)nb * 512 * 1024;
    const float* bfbf = fbf + (size_t)nb * 1024;

    layernorm_kernel<<<MT, 256, 0, stream>>>(X, X1, nullptr, nullptr);
    launch_gemm(1, false, false, X1, 1024, bdW1, bdb1, nullptr, 0, H1, 512, MP, 512, 1024, 1024, stream);
    launch_gemm(0, false, false, H1, 512, bdW2, bdb2, nullptr, 0, HD, 1024, MP, 1024, 512, 512, stream);
    // mamba
    launch_gemm(0, false, false, HD, 1024, binw, nullptr, nullptr, 0, XZ, 4096, MP, 4096, 1024, 1024, stream);
    dwconv_silu_kernel<<<grd(4 * 257 * 2048), 256, 0, stream>>>(XZ, bcw, bcb, XC);
    launch_gemm(0, false, true, XC, 2048, bxp, nullptr, nullptr, 0, DBC, 96, MP, 96, 2048, 2048, stream);
    launch_gemm(2, false, false, DBC, 96, bdtw, bdtb, nullptr, 0, DELTA, 2048, MP, 2048, 64, 64, stream);
    mamba_scan_kernel<<<dim3(8, 4), 256, 0, stream>>>(DELTA, DBC, XC, XZ, bAl, bDp, Yb);
    launch_gemm(0, false, false, Yb, 2048, bow, nullptr, nullptr, 0, MO, 1024, MP, 1024, 2048, 2048, stream);
    // gated fuse: RO = silu(X1@fWr+fbr); AO = silu(MO@fWm+fbm) * RO  (fused epilogue)
    launch_gemm(1, false, false, X1, 1024, bfWr, bfbr, nullptr, 0, RO, 512, MP, 512, 1024, 1024, stream);
    launch_gemm(3, false, false, MO, 1024, bfWm, bfbm, RO, 512, AO, 512, MP, 512, 1024, 1024, stream);
    launch_gemm(0, true, false, AO, 512, bfWf, bfbf, X, 1024, X, 1024, MP, 1024, 512, 512, stream);
  }

  // --- head ---
  layernorm_kernel<<<MT, 256, 0, stream>>>(X, XF, gamma, beta);
  launch_gemm(0, false, true, XF, 1024, dec_w, dec_b, nullptr, 0, DEC, 16, MP, 16, 1024, 1024, stream);
  final_head_kernel<<<grd(16384), 256, 0, stream>>>(DEC, fconv_w, fconv_b, out);
}
